// COSSIMMLP_44023414784013
// MI455X (gfx1250) — compile-verified, matrix operand order checked by reading, not run on hardware
//
#include <hip/hip_runtime.h>
#include <hip/hip_bf16.h>
#include <math.h>

// Fixed problem shape from the reference.
#define B_ 4
#define N_ 4096
#define D_ 256
#define E_ 131072
#define EPS_ 1e-8f

typedef __attribute__((ext_vector_type(16))) _Float16 v16h;
typedef __attribute__((ext_vector_type(8)))  float    v8f;
typedef __attribute__((ext_vector_type(4)))  float    v4f;

// ---------------------------------------------------------------------------
// Kernel 1: zero the 256 MB adjacency with nontemporal 128-bit stores.
// Pure store-bandwidth; this is the HBM roofline term (~11 us at 23.3 TB/s).
// ---------------------------------------------------------------------------
__global__ void cossim_zero_kernel(v4f* __restrict__ out, long n4) {
    long idx    = (long)blockIdx.x * blockDim.x + threadIdx.x;
    long stride = (long)gridDim.x * blockDim.x;
    v4f z = {0.0f, 0.0f, 0.0f, 0.0f};
    for (long i = idx; i < n4; i += stride) {
        __builtin_nontemporal_store(z, &out[i]);
    }
}

// ---------------------------------------------------------------------------
// Kernel 2: per-row inverse norms. One wave32 per row (B*N = 16384 rows).
// Each lane loads 8 contiguous floats (2x B128), squares, xor-shuffle reduce.
// invn[row] = 1 / max(||row||, EPS)
// ---------------------------------------------------------------------------
__global__ void cossim_norms_kernel(const float* __restrict__ prop,
                                    float* __restrict__ invn) {
    const int lane = threadIdx.x & 31;
    const int row  = ((int)blockIdx.x * blockDim.x + threadIdx.x) >> 5; // wave id
    const float* r = prop + (long)row * D_;

    v4f a = *(const v4f*)(r + lane * 8);
    v4f b = *(const v4f*)(r + lane * 8 + 4);
    float ss = a.x * a.x + a.y * a.y + a.z * a.z + a.w * a.w
             + b.x * b.x + b.y * b.y + b.z * b.z + b.w * b.w;

    #pragma unroll
    for (int m = 16; m >= 1; m >>= 1)
        ss += __shfl_xor(ss, m, 32);

    if (lane == 0) {
        float nrm = fmaxf(sqrtf(ss), EPS_);
        invn[row] = 1.0f / nrm;
    }
}

// ---------------------------------------------------------------------------
// Kernel 3: 16 edges per wave via WMMA.
//   A tile  = 16 gathered src rows   (16x32 f16 per step, ISA A layout)
//   B tile  = 16 gathered dst rows   (32x16, column-major == same lane layout)
//   acc     = A * B^T accumulated over K=256 in 8 wmma_f32_16x16x32_f16 steps
// Edge e's dot product is the diagonal element acc D[e][e]:
//   lanes 0..7  hold D[l][l]     in acc[l]
//   lanes 24..31 hold D[l-16][l-16] in acc[l-24]
// Then val = sigmoid(dot * invn[i] * invn[j]), scattered symmetrically.
//
// ISA 16-bit A layout (16x32): lane l<16 handles row l, lane l>=16 handles
// row l-16; half-vector elems 0..7 = K (off..off+7), elems 8..15 =
// K (16+off..16+off+7), where off = (l>=16) ? 8 : 0.
// ---------------------------------------------------------------------------
__global__ void cossim_edges_kernel(const float* __restrict__ prop,
                                    const int*   __restrict__ mask,
                                    const float* __restrict__ invn,
                                    float* __restrict__ adj) {
    const int lane = threadIdx.x & 31;
    const int wave = ((int)blockIdx.x * blockDim.x + threadIdx.x) >> 5;

    const int tilesPerBatch = E_ / 16;                 // 8192
    const int b  = wave / tilesPerBatch;
    const int e0 = (wave % tilesPerBatch) * 16;
    const int el = lane & 15;                          // edge handled by lane
    const int e  = e0 + el;

    const long mbase = (((long)b * E_ + e) << 1);
    const int i = mask[mbase + 0];
    const int j = mask[mbase + 1];

    const float* srow = prop + ((long)b * N_ + i) * D_;
    const float* drow = prop + ((long)b * N_ + j) * D_;
    const int off = (lane >> 4) ? 8 : 0;

    v8f acc = {0.f, 0.f, 0.f, 0.f, 0.f, 0.f, 0.f, 0.f};

    #pragma unroll
    for (int kb = 0; kb < D_; kb += 32) {
        // --- A tile (src rows), per-lane 16 contiguous-in-pairs f32 -> f16
        v4f s0 = *(const v4f*)(srow + kb + off);
        v4f s1 = *(const v4f*)(srow + kb + off + 4);
        v4f s2 = *(const v4f*)(srow + kb + off + 16);
        v4f s3 = *(const v4f*)(srow + kb + off + 20);
        // --- B tile (dst rows, column-major => identical lane layout)
        v4f t0 = *(const v4f*)(drow + kb + off);
        v4f t1 = *(const v4f*)(drow + kb + off + 4);
        v4f t2 = *(const v4f*)(drow + kb + off + 16);
        v4f t3 = *(const v4f*)(drow + kb + off + 20);

        v16h A, Bm;
        A[0]  = (_Float16)s0.x;  A[1]  = (_Float16)s0.y;
        A[2]  = (_Float16)s0.z;  A[3]  = (_Float16)s0.w;
        A[4]  = (_Float16)s1.x;  A[5]  = (_Float16)s1.y;
        A[6]  = (_Float16)s1.z;  A[7]  = (_Float16)s1.w;
        A[8]  = (_Float16)s2.x;  A[9]  = (_Float16)s2.y;
        A[10] = (_Float16)s2.z;  A[11] = (_Float16)s2.w;
        A[12] = (_Float16)s3.x;  A[13] = (_Float16)s3.y;
        A[14] = (_Float16)s3.z;  A[15] = (_Float16)s3.w;

        Bm[0]  = (_Float16)t0.x; Bm[1]  = (_Float16)t0.y;
        Bm[2]  = (_Float16)t0.z; Bm[3]  = (_Float16)t0.w;
        Bm[4]  = (_Float16)t1.x; Bm[5]  = (_Float16)t1.y;
        Bm[6]  = (_Float16)t1.z; Bm[7]  = (_Float16)t1.w;
        Bm[8]  = (_Float16)t2.x; Bm[9]  = (_Float16)t2.y;
        Bm[10] = (_Float16)t2.z; Bm[11] = (_Float16)t2.w;
        Bm[12] = (_Float16)t3.x; Bm[13] = (_Float16)t3.y;
        Bm[14] = (_Float16)t3.z; Bm[15] = (_Float16)t3.w;

        // 8 args: (neg_a, A, neg_b, B, c_mod, C, reuse_a, reuse_b)
        acc = __builtin_amdgcn_wmma_f32_16x16x32_f16(
            false, A, false, Bm, (short)0, acc, false, false);
    }

    // Extract diagonal: lane<8 -> acc[lane]; lane>=24 -> acc[lane-24].
    const bool active = (lane < 8) || (lane >= 24);
    const int  r = (lane < 8) ? lane : (lane - 24);
    float dot = acc[0];
    if (r == 1) dot = acc[1];
    if (r == 2) dot = acc[2];
    if (r == 3) dot = acc[3];
    if (r == 4) dot = acc[4];
    if (r == 5) dot = acc[5];
    if (r == 6) dot = acc[6];
    if (r == 7) dot = acc[7];

    if (active) {
        float v = dot * invn[b * N_ + i] * invn[b * N_ + j];
        float s = 1.0f / (1.0f + __expf(-v));
        adj[((long)b * N_ + i) * N_ + j] = s;
        adj[((long)b * N_ + j) * N_ + i] = s;
    }
}

// ---------------------------------------------------------------------------
extern "C" void kernel_launch(void* const* d_in, const int* in_sizes, int n_in,
                              void* d_out, int out_size, void* d_ws, size_t ws_size,
                              hipStream_t stream) {
    (void)in_sizes; (void)n_in; (void)ws_size;

    const float* prop = (const float*)d_in[0];
    const int*   mask = (const int*)d_in[1];
    float* adj  = (float*)d_out;
    float* invn = (float*)d_ws;                 // B_*N_ floats = 64 KB scratch

    // 1) zero 256 MB output (store-bandwidth bound)
    long n4 = (long)out_size / 4;               // float4 count
    cossim_zero_kernel<<<8192, 256, 0, stream>>>((v4f*)adj, n4);

    // 2) inverse norms: B*N = 16384 rows, 8 waves/block -> 2048 blocks
    cossim_norms_kernel<<<(B_ * N_) / 8, 256, 0, stream>>>(prop, invn);

    // 3) edges: B*E/16 = 32768 waves, 8 waves/block -> 4096 blocks (exact)
    cossim_edges_kernel<<<(B_ * E_ / 16) / 8, 256, 0, stream>>>(prop, mask, invn, adj);
}
